// MoELayer_13932873908550
// MI455X (gfx1250) — matrix operand order repaired and checked by reading, not compiled, not run on hardware
//
#include <hip/hip_runtime.h>
#include <hip/hip_bf16.h>
#include <math.h>

// ---------------------------------------------------------------------------
// MoE layer for MI455X (gfx1250): bf16 WMMA 16x16x32, f32 accumulate.
// Sizes fixed by the reference: B=2,S=4096 -> T=8192, D=512, H=2048, E=8, K=2
// ---------------------------------------------------------------------------

#define TT 8192
#define DD 512
#define HH 2048
#define EE 8
#define MT 64            // tokens per workgroup in main kernel
#define LDX 520          // padded LDS row stride (bf16 elems) for x tile
#define LDH 72           // padded LDS row stride (bf16 elems) for h chunk

typedef __attribute__((ext_vector_type(16))) __bf16 v16bf;
typedef __attribute__((ext_vector_type(8)))  __bf16 v8bf;
typedef __attribute__((ext_vector_type(8)))  float  v8f;
typedef __attribute__((ext_vector_type(4)))  int    v4i;

__device__ __forceinline__ v8f wmma_bf16(v16bf a, v16bf b, v8f c) {
  // D = A(16x32) * B(32x16) + C, f32 accumulate
  return __builtin_amdgcn_wmma_f32_16x16x32_bf16(
      /*neg_a=*/false, a, /*neg_b=*/false, b,
      /*c_mod=*/(short)0, c, /*reuse_a=*/false, /*reuse_b=*/false);
}

// A operand as this code loads it (16x32 bf16, contiguous k order, B matches): lane L holds row M=(L&15);
// 16 contiguous elems at K = kbase..kbase+15, kbase = k0 + 16*(L>=16).
__device__ __forceinline__ v16bf load_A(const __bf16* p) {
  v8bf lo = *(const v8bf*)(p);
  v8bf hi = *(const v8bf*)(p + 8);
  return __builtin_shufflevector(lo, hi, 0,1,2,3,4,5,6,7,8,9,10,11,12,13,14,15);
}

__device__ __forceinline__ float gelu_exact(float x) {
  return 0.5f * x * (1.0f + erff(x * 0.70710678118654752f));
}

// ---------------------------------------------------------------------------
// Kernel 1: router. One wave per token. gate[t][e] = top2-masked softmax.
// ---------------------------------------------------------------------------
__global__ void router_kernel(const float* __restrict__ x,
                              const float* __restrict__ gw,
                              const float* __restrict__ gb,
                              float* __restrict__ gate) {
  int wave = threadIdx.x >> 5;
  int lane = threadIdx.x & 31;
  int t = blockIdx.x * 8 + wave;

  float acc[EE];
#pragma unroll
  for (int e = 0; e < EE; ++e) acc[e] = 0.0f;

  const float* xr = x + (size_t)t * DD;
  for (int d = lane; d < DD; d += 32) {
    float xv = xr[d];
    const float* g = gw + (size_t)d * EE;
#pragma unroll
    for (int e = 0; e < EE; ++e) acc[e] += xv * g[e];
  }
#pragma unroll
  for (int off = 16; off > 0; off >>= 1) {
#pragma unroll
    for (int e = 0; e < EE; ++e) acc[e] += __shfl_xor(acc[e], off, 32);
  }
#pragma unroll
  for (int e = 0; e < EE; ++e) acc[e] += gb[e];

  float m = acc[0];
#pragma unroll
  for (int e = 1; e < EE; ++e) m = fmaxf(m, acc[e]);
  float p[EE], s = 0.0f;
#pragma unroll
  for (int e = 0; e < EE; ++e) { p[e] = expf(acc[e] - m); s += p[e]; }
  float inv = 1.0f / s;
#pragma unroll
  for (int e = 0; e < EE; ++e) p[e] *= inv;

  // top-2 (first-occurrence tie break, like jax top_k)
  int i1 = 0;
#pragma unroll
  for (int e = 1; e < EE; ++e) if (p[e] > p[i1]) i1 = e;
  int i2 = (i1 == 0) ? 1 : 0;
#pragma unroll
  for (int e = 0; e < EE; ++e) if (e != i1 && p[e] > p[i2]) i2 = e;

  if (lane < EE) {
    float v = (lane == i1) ? p[i1] : ((lane == i2) ? p[i2] : 0.0f);
    gate[(size_t)t * EE + lane] = v;
  }
}

// ---------------------------------------------------------------------------
// Kernel 2: x (f32) -> xb (bf16), 4 elems/thread
// ---------------------------------------------------------------------------
__global__ void cvt_x_kernel(const float* __restrict__ x, __bf16* __restrict__ xb) {
  int i = (blockIdx.x * 256 + threadIdx.x) * 4;
  float4 v = *(const float4*)(x + i);
  xb[i + 0] = (__bf16)v.x;
  xb[i + 1] = (__bf16)v.y;
  xb[i + 2] = (__bf16)v.z;
  xb[i + 3] = (__bf16)v.w;
}

// ---------------------------------------------------------------------------
// Kernel 3: transpose last two dims + convert to bf16.
// src: [E][R][C] f32 -> dst: [E][C][R] bf16.  grid (C/32, R/32, E), block 32x8
// ---------------------------------------------------------------------------
__global__ void transpose_cvt_kernel(const float* __restrict__ src,
                                     __bf16* __restrict__ dst,
                                     int R, int C) {
  __shared__ float tile[32][33];
  int e  = blockIdx.z;
  int c0 = blockIdx.x * 32;
  int r0 = blockIdx.y * 32;
  const float* s = src + (size_t)e * R * C;
  __bf16*      d = dst + (size_t)e * R * C;
  int tx = threadIdx.x, ty = threadIdx.y;
#pragma unroll
  for (int i = 0; i < 32; i += 8)
    tile[ty + i][tx] = s[(size_t)(r0 + ty + i) * C + (c0 + tx)];
  __syncthreads();
#pragma unroll
  for (int i = 0; i < 32; i += 8)
    d[(size_t)(c0 + ty + i) * R + (r0 + tx)] = (__bf16)tile[tx][ty + i];
}

// ---------------------------------------------------------------------------
// Kernel 4: fused MoE FFN.
//   xb  : [T][D]  bf16 (row-major, K-contiguous A for GEMM1)
//   w1t : [E][H][D] bf16 (column h has K=d contiguous -> B for GEMM1)
//   w2t : [E][D][H] bf16 (column d has K=h contiguous -> B for GEMM2)
//   out[t][:] = sum_e  (g[t,e]*gelu(x@w1[e]+b1[e])) @ w2[e]  + sum_e g[t,e]*b2[e]
// ---------------------------------------------------------------------------
__global__ __launch_bounds__(256) void moe_main_kernel(
    const __bf16* __restrict__ xb, const __bf16* __restrict__ w1t,
    const __bf16* __restrict__ w2t, const float* __restrict__ gate,
    const float* __restrict__ b1, const float* __restrict__ b2,
    float* __restrict__ out) {
  __shared__ __bf16 xs[MT * LDX];   // 65 KB  x tile
  __shared__ __bf16 hs[MT * LDH];   //  9 KB  current h chunk (gated, bf16)
  __shared__ float  gs[MT * EE];    //  2 KB  gates for this tile

  const int tid  = threadIdx.x;
  const int wave = tid >> 5;
  const int lane = tid & 31;
  const int l15  = lane & 15;
  const int lhi  = lane >> 4;           // 0 for lanes 0-15, 1 for 16-31
  const int tok0 = blockIdx.x * MT;

  // ---- stage x tile into LDS (async DMA path if available) ----
#if __has_builtin(__builtin_amdgcn_global_load_async_to_lds_b128)
  for (int c = tid; c < MT * 64; c += 256) {     // 64 x 16B-chunks per row
    int row = c >> 6, col = (c & 63) << 3;
    __builtin_amdgcn_global_load_async_to_lds_b128(
        (v4i*)&xb[(size_t)(tok0 + row) * DD + col],
        (v4i*)&xs[row * LDX + col],
        0, 0);
  }
#else
  for (int c = tid; c < MT * 64; c += 256) {
    int row = c >> 6, col = (c & 63) << 3;
    *(v8bf*)&xs[row * LDX + col] =
        *(const v8bf*)&xb[(size_t)(tok0 + row) * DD + col];
  }
#endif
  for (int i = tid; i < MT * EE; i += 256) gs[i] = gate[(size_t)tok0 * EE + i];
#if __has_builtin(__builtin_amdgcn_s_wait_asynccnt)
  __builtin_amdgcn_s_wait_asynccnt(0);
#else
  asm volatile("s_wait_asynccnt 0" ::: "memory");
#endif
  __syncthreads();

  // GEMM1 tile assignment: wave -> rows [16*(w>>1)), 2 adjacent 16-col tiles
  const int m0a = (wave >> 1) * 16;
  const int nc1 = (wave & 1) * 2;
  // GEMM2 / output assignment: wave -> rows 16*(w&3), d range (w>>2)*256..+255
  const int m0b   = (wave & 3) * 16;
  const int dbase = (wave >> 2) * 256;

  v8f acc[16];
#pragma unroll
  for (int i = 0; i < 16; ++i) { v8f z = {}; acc[i] = z; }

  for (int e = 0; e < EE; ++e) {
    // gate values for this wave's GEMM1 row set (C-layout: M = j + 8*lhi)
    float gv[8];
#pragma unroll
    for (int j = 0; j < 8; ++j) gv[j] = gs[(m0a + j + 8 * lhi) * EE + e];

    for (int hc = 0; hc < HH / 64; ++hc) {
      // ---------------- GEMM1: h chunk [MT x 64] ----------------
      v8f c0 = {}, c1 = {};
      const int hcol0 = hc * 64 + nc1 * 16 + l15;  // this lane's h column, tile0
      const __bf16* wB0 = w1t + ((size_t)e * HH + hcol0) * DD;
      const __bf16* wB1 = wB0 + (size_t)16 * DD;   // tile1: +16 h columns
      const __bf16* arow = &xs[(m0a + l15) * LDX + 16 * lhi];

      // pull this k-stream into the WGP-near caches (locality 3 -> WGP scope;
      // speculative prefetch, drop-on-OOB safe)
#pragma unroll
      for (int pf = 0; pf < DD; pf += 128) {
        __builtin_prefetch(wB0 + pf, 0, 3);
        __builtin_prefetch(wB1 + pf, 0, 3);
      }

#pragma unroll 4
      for (int k0 = 0; k0 < DD; k0 += 32) {
        v16bf a  = load_A(arow + k0);
        v16bf b0 = *(const v16bf*)(wB0 + k0 + 16 * lhi);
        v16bf b1v = *(const v16bf*)(wB1 + k0 + 16 * lhi);
        c0 = wmma_bf16(a, b0, c0);
        c1 = wmma_bf16(a, b1v, c1);
      }
      const float bb0 = b1[(size_t)e * HH + hc * 64 + nc1 * 16 + l15];
      const float bb1 = b1[(size_t)e * HH + hc * 64 + nc1 * 16 + 16 + l15];
#pragma unroll
      for (int j = 0; j < 8; ++j) {
        float v0 = gelu_exact(c0[j] + bb0) * gv[j];
        float v1 = gelu_exact(c1[j] + bb1) * gv[j];
        int r = m0a + j + 8 * lhi;
        hs[r * LDH + nc1 * 16 + l15]      = (__bf16)v0;
        hs[r * LDH + nc1 * 16 + 16 + l15] = (__bf16)v1;
      }
      __syncthreads();

      // prefetch next h-chunk's slice of the w2 k-stream (WGP scope)
#pragma unroll
      for (int nt = 0; nt < 16; nt += 4) {
        int d = dbase + nt * 16 + l15;
        __builtin_prefetch(w2t + ((size_t)e * DD + d) * HH + hc * 64 + 64, 0, 3);
      }

      // ---------------- GEMM2: acc += h_chunk @ w2t ----------------
#pragma unroll
      for (int kk = 0; kk < 2; ++kk) {
        const int k0 = kk * 32;
        v16bf a = load_A(&hs[(m0b + l15) * LDH + k0 + 16 * lhi]);
        const int kg = hc * 64 + k0 + 16 * lhi;
#pragma unroll
        for (int nt = 0; nt < 16; ++nt) {
          int d = dbase + nt * 16 + l15;
          v16bf b = *(const v16bf*)(w2t + ((size_t)e * DD + d) * HH + kg);
          acc[nt] = wmma_bf16(a, b, acc[nt]);
        }
      }
      __syncthreads();   // hs will be overwritten next chunk
    }
  }

  // ---- epilogue: add sum_e g[t,e]*b2[e][d], store f32 output ----
#pragma unroll
  for (int nt = 0; nt < 16; ++nt) {
    int d = dbase + nt * 16 + l15;
    float bt[EE];
#pragma unroll
    for (int e = 0; e < EE; ++e) bt[e] = b2[(size_t)e * DD + d];
#pragma unroll
    for (int j = 0; j < 8; ++j) {
      int m = m0b + j + 8 * lhi;
      float v = acc[nt][j];
#pragma unroll
      for (int e = 0; e < EE; ++e) v += gs[m * EE + e] * bt[e];
      out[(size_t)(tok0 + m) * DD + d] = v;
    }
  }
}

// ---------------------------------------------------------------------------
// Host launcher
// ---------------------------------------------------------------------------
extern "C" void kernel_launch(void* const* d_in, const int* in_sizes, int n_in,
                              void* d_out, int out_size, void* d_ws, size_t ws_size,
                              hipStream_t stream) {
  const float* x  = (const float*)d_in[0];   // [2,4096,512]
  const float* gw = (const float*)d_in[1];   // [512,8]
  const float* gb = (const float*)d_in[2];   // [8]
  const float* w1 = (const float*)d_in[3];   // [8,512,2048]
  const float* b1 = (const float*)d_in[4];   // [8,2048]
  const float* w2 = (const float*)d_in[5];   // [8,2048,512]
  const float* b2 = (const float*)d_in[6];   // [8,512]
  float* out = (float*)d_out;                // [2,4096,512]

  // workspace layout
  char* ws = (char*)d_ws;
  float*  gate = (float*)ws;                                   // T*E f32
  size_t off = (size_t)TT * EE * sizeof(float);
  __bf16* xbuf = (__bf16*)(ws + off);                          // T*D bf16
  off += (size_t)TT * DD * sizeof(__bf16);
  __bf16* w1t = (__bf16*)(ws + off);                           // E*H*D bf16
  off += (size_t)EE * HH * DD * sizeof(__bf16);
  __bf16* w2t = (__bf16*)(ws + off);                           // E*D*H bf16

  router_kernel<<<TT / 8, 256, 0, stream>>>(x, gw, gb, gate);
  cvt_x_kernel<<<(TT * DD) / (256 * 4), 256, 0, stream>>>(x, xbuf);

  dim3 tb(32, 8);
  // w1 [E][D][H] -> w1t [E][H][D]
  transpose_cvt_kernel<<<dim3(HH / 32, DD / 32, EE), tb, 0, stream>>>(w1, w1t, DD, HH);
  // w2 [E][H][D] -> w2t [E][D][H]
  transpose_cvt_kernel<<<dim3(DD / 32, HH / 32, EE), tb, 0, stream>>>(w2, w2t, HH, DD);

  moe_main_kernel<<<TT / MT, 256, 0, stream>>>(xbuf, w1t, w2t, gate, b1, b2, out);
}
